// _SelfAttentiveLBLEncoder_39367670235446
// MI455X (gfx1250) — matrix operand order repaired and checked by reading, not compiled, not run on hardware
//
#include <hip/hip_runtime.h>
#include <hip/hip_bf16.h>

// ---------------------------------------------------------------------------
// SelfAttentiveLBLEncoder for MI455X (gfx1250, wave32, WMMA).
// L=2 layers x 2 directions; HID=512, H=8 heads (dk=64), B=4, S=1024,
// band width W+1 = 34 (W = 33). All GEMMs in bf16 with f32 accumulation via
// v_wmma_f32_16x16x32_bf16; GEMM global->LDS staging uses the CDNA5 async
// path (global_load_async_to_lds_b128 + s_wait_asynccnt) with double-buffered
// LDS tiles; softmax / highway gate / residual in f32.
// ---------------------------------------------------------------------------

typedef __attribute__((ext_vector_type(16))) __bf16 v16bf;
typedef __attribute__((ext_vector_type(8)))  float  v8f;

union Frag16 { v16bf v; uint4 q[2]; };

__device__ __forceinline__ __bf16 f2bf(float f) {
  union { float f; unsigned u; } a; a.f = f;
  unsigned r = a.u + 0x7fffu + ((a.u >> 16) & 1u);   // round to nearest even
  unsigned short h = (unsigned short)(r >> 16);
  union { unsigned short s; __bf16 b; } o; o.s = h;
  return o.b;
}

// CDNA5 async copy: global memory -> LDS, 16B per lane, tracked by ASYNCcnt.
// VDST = LDS byte address (low 32 bits of the generic pointer; LDS aperture
// addresses truncate to addr[31:0] per ISA 10.2), VADDR = 64-bit global addr.
__device__ __forceinline__ void async_ld_b128(void* lds, const void* gptr) {
  unsigned l = (unsigned)(uintptr_t)lds;
  asm volatile("global_load_async_to_lds_b128 %0, %1, off"
               :: "v"(l), "v"(gptr) : "memory");
}

// ---------------------------------------------------------------------------
// cast f32 -> bf16
// ---------------------------------------------------------------------------
__global__ void cast_f32_bf16(const float* __restrict__ in,
                              __bf16* __restrict__ out, int n) {
  int i = blockIdx.x * blockDim.x + threadIdx.x;
  if (i < n) out[i] = f2bf(in[i]);
}

// ---------------------------------------------------------------------------
// GEMM: C[M,N] = A_bf16[M,K] @ W_bf16[N,K]^T + bias[N]
// (torch Linear: weight is [out,in] so both operands are K-contiguous)
// Block 128 threads = 4 waves; 64x64 tile; each wave 32x32 (4 WMMA accums).
// Double-buffered LDS tiles filled by async global->LDS loads.
// M,N multiples of 64; K multiple of 32 (here M=4096, N in {512,1024}, K=512).
// ---------------------------------------------------------------------------
#define LDSTR 40   // 32 K-elements + 8 pad (bf16) -> conflict-free b128 reads

__global__ __launch_bounds__(128)
void gemm_bf16_wmma(const __bf16* __restrict__ A, const __bf16* __restrict__ Bw,
                    const float* __restrict__ bias,
                    float* __restrict__ Cf, __bf16* __restrict__ Cb,
                    int M, int N, int K) {
  __shared__ __bf16 lA[2][64 * LDSTR];
  __shared__ __bf16 lB[2][64 * LDSTR];
  const int tid  = threadIdx.x;
  const int lane = tid & 31;
  const int wave = tid >> 5;            // 0..3
  const int wm = wave >> 1, wn = wave & 1;
  const int half = lane >> 4;           // 0/1
  const int lrow = lane & 15;
  const int n0 = blockIdx.x * 64;
  const int m0 = blockIdx.y * 64;

  // issue one 64x32 A-tile + 64x32 W-tile as async b128 copies
  // (256 chunks of 8 bf16 per tile; 2 chunks per thread per tile
  //  -> 4 async instructions per wave per k-step)
  auto issue = [&](int buf, int k0) {
#pragma unroll
    for (int i = 0; i < 2; ++i) {
      int c = i * 128 + tid;            // 0..255
      int row = c >> 2, col8 = (c & 3) * 8;
      async_ld_b128(&lA[buf][row * LDSTR + col8],
                    &A[(size_t)(m0 + row) * K + k0 + col8]);
      async_ld_b128(&lB[buf][row * LDSTR + col8],
                    &Bw[(size_t)(n0 + row) * K + k0 + col8]);
    }
  };

  v8f acc[2][2] = {};
  const int nt = K / 32;
  issue(0, 0);
  for (int it = 0; it < nt; ++it) {
    const int cur = it & 1;
    if (it + 1 < nt) {
      issue(cur ^ 1, (it + 1) * 32);
      // async ops complete in order: <=4 outstanding means the current
      // tile's 4 per-wave copies have landed in LDS
      asm volatile("s_wait_asynccnt 0x4" ::: "memory");
    } else {
      asm volatile("s_wait_asynccnt 0x0" ::: "memory");
    }
    __syncthreads();

    // A fragment (16x32): lane holds row M=lrow; K chunks {half*8..},{16+half*8..}
    Frag16 a[2], b[2];
#pragma unroll
    for (int f = 0; f < 2; ++f) {
      int row = wm * 32 + f * 16 + lrow;
      a[f].q[0] = *(const uint4*)&lA[cur][row * LDSTR + half * 8];
      a[f].q[1] = *(const uint4*)&lA[cur][row * LDSTR + 16 + half * 8];
    }
    // B fragment (32x16): lane (half,n=lrow) holds W[n][k0+half*8..+7], W[n][k0+16+half*8..+7]
#pragma unroll
    for (int g = 0; g < 2; ++g) {
      int row = wn * 32 + g * 16 + lrow;
      b[g].q[0] = *(const uint4*)&lB[cur][row * LDSTR + half * 8];
      b[g].q[1] = *(const uint4*)&lB[cur][row * LDSTR + 16 + half * 8];
    }
#pragma unroll
    for (int f = 0; f < 2; ++f)
#pragma unroll
      for (int g = 0; g < 2; ++g)
        acc[f][g] = __builtin_amdgcn_wmma_f32_16x16x32_bf16(
            false, a[f].v, false, b[g].v, (short)0, acc[f][g], false, false);
    __syncthreads();   // all waves done reading buf[cur] before it is refilled
  }

  // epilogue: D element (lane,r) -> M = r + 8*half, N = lrow
#pragma unroll
  for (int f = 0; f < 2; ++f)
#pragma unroll
    for (int g = 0; g < 2; ++g) {
      int nglob = n0 + wn * 32 + g * 16 + lrow;
      float bv = bias ? bias[nglob] : 0.f;
#pragma unroll
      for (int r = 0; r < 8; ++r) {
        int mglob = m0 + wm * 32 + f * 16 + r + half * 8;
        float v = acc[f][g][r] + bv;
        if (Cf) Cf[(size_t)mglob * N + nglob] = v;
        if (Cb) Cb[(size_t)mglob * N + nglob] = f2bf(v);
      }
    }
}

// ---------------------------------------------------------------------------
// Banded relative-position attention. One wave per 16-query tile per (b,h).
// Q,K,V bf16 [B*S, 512]; head h at columns h*64. Band: left: j in [q-33,q];
// right: j in [q,q+33]. 4 key tiles of 16 cover band + tile skew.
// ---------------------------------------------------------------------------
__global__ __launch_bounds__(128)
void attn_banded(const __bf16* __restrict__ Q, const __bf16* __restrict__ Kd,
                 const __bf16* __restrict__ V, const float* __restrict__ rel,
                 __bf16* __restrict__ O, int left) {
  __shared__ __bf16 sP[4][16 * 72];    // P[q][key] bf16, stride 72
  __shared__ __bf16 sVt[4][64 * 72];   // V^T[c][key] bf16, stride 72
  const int tid = threadIdx.x, lane = tid & 31, w = tid >> 5;
  const int wid = blockIdx.x * 4 + w;
  const int qt = wid & 63;             // S/16 = 64 query tiles
  const int h  = (wid >> 6) & 7;
  const int b  = wid >> 9;
  const int half = lane >> 4, lrow = lane & 15;
  const int q0 = qt * 16;
  const size_t rowbase = (size_t)b * 1024;
  const int col0 = h * 64;
  __bf16* P  = sP[w];
  __bf16* Vt = sVt[w];

  // Q A-fragments: 2 chunks of K(feature)=32
  Frag16 qa[2];
  {
    size_t qrow = (rowbase + q0 + lrow) * 512 + col0;
#pragma unroll
    for (int c = 0; c < 2; ++c) {
      qa[c].q[0] = *(const uint4*)&Q[qrow + c * 32 + half * 8];
      qa[c].q[1] = *(const uint4*)&Q[qrow + c * 32 + 16 + half * 8];
    }
  }

  const int jbase = left ? (q0 - 48) : q0;
  v8f sc[4];
#pragma unroll
  for (int kt = 0; kt < 4; ++kt) {
    int j = jbase + kt * 16 + lrow;
    int jc = min(max(j, 0), 1023);     // clamp; invalid keys masked below
    size_t krow = (rowbase + jc) * 512 + col0;
    v8f s = {};
#pragma unroll
    for (int c = 0; c < 2; ++c) {
      Frag16 kb;
      kb.q[0] = *(const uint4*)&Kd[krow + c * 32 + half * 8];
      kb.q[1] = *(const uint4*)&Kd[krow + c * 32 + 16 + half * 8];
      s = __builtin_amdgcn_wmma_f32_16x16x32_bf16(false, qa[c].v, false, kb.v,
                                                  (short)0, s, false, false);
    }
    sc[kt] = s;
  }

  // bias + band mask + softmax (rows live across 16 lanes of a half)
  const float scale = 0.125f;          // 1/sqrt(64)
  float rmax[8];
#pragma unroll
  for (int r = 0; r < 8; ++r) rmax[r] = -3.0e38f;
#pragma unroll
  for (int kt = 0; kt < 4; ++kt) {
    int j = jbase + kt * 16 + lrow;
#pragma unroll
    for (int r = 0; r < 8; ++r) {
      int q = q0 + r + half * 8;
      int off = left ? (q - j) : (j - q);
      float s;
      if (off >= 0 && off <= 33 && j >= 0 && j < 1024) {
        int idx = left ? (33 - off) : off;
        s = sc[kt][r] * scale + rel[h * 34 + idx];
      } else {
        s = -1.0e9f;
      }
      sc[kt][r] = s;
      rmax[r] = fmaxf(rmax[r], s);
    }
  }
#pragma unroll
  for (int r = 0; r < 8; ++r) {
    float m = rmax[r];
    m = fmaxf(m, __shfl_xor(m, 1, 32));
    m = fmaxf(m, __shfl_xor(m, 2, 32));
    m = fmaxf(m, __shfl_xor(m, 4, 32));
    m = fmaxf(m, __shfl_xor(m, 8, 32));
    rmax[r] = m;
  }
  float rsum[8] = {};
#pragma unroll
  for (int kt = 0; kt < 4; ++kt)
#pragma unroll
    for (int r = 0; r < 8; ++r) {
      float e = __expf(sc[kt][r] - rmax[r]);
      sc[kt][r] = e;
      rsum[r] += e;
    }
#pragma unroll
  for (int r = 0; r < 8; ++r) {
    float s = rsum[r];
    s += __shfl_xor(s, 1, 32);
    s += __shfl_xor(s, 2, 32);
    s += __shfl_xor(s, 4, 32);
    s += __shfl_xor(s, 8, 32);
    rsum[r] = 1.0f / s;
  }
  // write P to LDS: P[q_local][key_local]
#pragma unroll
  for (int kt = 0; kt < 4; ++kt)
#pragma unroll
    for (int r = 0; r < 8; ++r)
      P[(r + half * 8) * 72 + kt * 16 + lrow] = f2bf(sc[kt][r] * rsum[r]);

  // stage V^T: Vt[c][key] for 64 key rows x 64 features
#pragma unroll
  for (int t = 0; t < 16; ++t) {
    int c = t * 32 + lane;             // 0..511 chunks of 8 bf16
    int keyrow = c >> 3, col8 = (c & 7) * 8;
    int j = jbase + keyrow;
    int jc = min(max(j, 0), 1023);
    uint4 d = *(const uint4*)&V[(rowbase + jc) * 512 + col0 + col8];
    const __bf16* e = (const __bf16*)&d;
#pragma unroll
    for (int x = 0; x < 8; ++x)
      Vt[(col8 + x) * 72 + keyrow] = e[x];
  }
  __syncthreads();

  // O^T = V^T @ P^T : 4 feature chunks x 2 key chunks of 32
  v8f oacc[4] = {};
#pragma unroll
  for (int kk = 0; kk < 2; ++kk) {
    Frag16 pb;  // B frag: lane (half,n=lrow) holds P[n][kk*32 + half*8 ..+7], P[n][kk*32 + 16 + half*8 ..+7]
    pb.q[0] = *(const uint4*)&P[lrow * 72 + kk * 32 + half * 8];
    pb.q[1] = *(const uint4*)&P[lrow * 72 + kk * 32 + 16 + half * 8];
#pragma unroll
    for (int ci = 0; ci < 4; ++ci) {
      Frag16 va;  // A frag: row = feature (ci*16+lrow), keys chunked
      int vr = (ci * 16 + lrow) * 72 + kk * 32;
      va.q[0] = *(const uint4*)&Vt[vr + half * 8];
      va.q[1] = *(const uint4*)&Vt[vr + 16 + half * 8];
      oacc[ci] = __builtin_amdgcn_wmma_f32_16x16x32_bf16(
          false, va.v, false, pb.v, (short)0, oacc[ci], false, false);
    }
  }
  // store: O[row = q0+lrow][col0 + ci*16 + half*8 + r] — 8 contiguous bf16
  size_t orow = (rowbase + q0 + lrow) * 512 + col0;
#pragma unroll
  for (int ci = 0; ci < 4; ++ci) {
    __bf16 tmp[8];
#pragma unroll
    for (int r = 0; r < 8; ++r) tmp[r] = f2bf(oacc[ci][r]);
    *(uint4*)&O[orow + ci * 16 + half * 8] = *(const uint4*)tmp;
  }
  __syncthreads();
}

// ---------------------------------------------------------------------------
// Highway gating: y = sigmoid(g)*x + (1-sigmoid(g))*relu(nl);
// proj [M,1024]: cols 0..511 = nonlin, 512..1023 = gate
// ---------------------------------------------------------------------------
__global__ void highway_elem(const float* __restrict__ proj,
                             const float* __restrict__ x,
                             float* __restrict__ yf, __bf16* __restrict__ yb,
                             int MN) {
  int i = blockIdx.x * blockDim.x + threadIdx.x;
  if (i >= MN) return;
  int m = i >> 9, c = i & 511;
  float nl = proj[(size_t)m * 1024 + c];
  float gt = proj[(size_t)m * 1024 + 512 + c];
  float g = 1.0f / (1.0f + __expf(-gt));
  float v = g * x[i] + (1.0f - g) * fmaxf(nl, 0.0f);
  yf[i] = v;
  yb[i] = f2bf(v);
}

// residual add + update x (f32+bf16) + write output slice (row stride 1024)
__global__ void residual_out(const float* __restrict__ h,
                             const float* __restrict__ cache, int use_cache,
                             float* __restrict__ xf, __bf16* __restrict__ xb,
                             float* __restrict__ outbase, int MN) {
  int i = blockIdx.x * blockDim.x + threadIdx.x;
  if (i >= MN) return;
  int m = i >> 9, c = i & 511;
  float v = h[i] + (use_cache ? cache[i] : 0.0f);
  xf[i] = v;
  xb[i] = f2bf(v);
  outbase[(size_t)m * 1024 + c] = v;
}

// ---------------------------------------------------------------------------
extern "C" void kernel_launch(void* const* d_in, const int* in_sizes, int n_in,
                              void* d_out, int out_size, void* d_ws, size_t ws_size,
                              hipStream_t stream) {
  const float* inputs = (const float*)d_in[0];
  // d_in[1] = masks (unused by the reference forward)
  const float* qkv_w = (const float*)d_in[2];  // [2,2,4,512,512]
  const float* qkv_b = (const float*)d_in[3];  // [2,2,4,512]
  const float* rel   = (const float*)d_in[4];  // [2,2,8,34]
  const float* hw_w  = (const float*)d_in[5];  // [2,2,2,1024,512]
  const float* hw_b  = (const float*)d_in[6];  // [2,2,2,1024]
  float* out = (float*)d_out;                  // [2,4,1024,1024]

  const int M = 4096;                          // B*S tokens
  const int HID = 512;
  const size_t ACT  = (size_t)M * HID;         // 2,097,152
  const size_t QKVW = (size_t)2 * 2 * 4 * 512 * 512;    // 4,194,304
  const size_t HWW  = (size_t)2 * 2 * 2 * 1024 * 512;   // 4,194,304

  char* ws = (char*)d_ws;
  size_t off = 0;
  auto alloc = [&](size_t bytes) -> void* {
    off = (off + 255) & ~(size_t)255;
    void* p = ws + off;
    off += bytes;
    return p;
  };

  __bf16* qkvw_bf = (__bf16*)alloc(QKVW * 2);
  __bf16* hww_bf  = (__bf16*)alloc(HWW * 2);
  float*  x_f[2];
  __bf16* x_b[2];
  float*  cachep[2];
  for (int dir = 0; dir < 2; ++dir) {
    x_f[dir]    = (float*)alloc(ACT * 4);
    x_b[dir]    = (__bf16*)alloc(ACT * 2);
    cachep[dir] = (float*)alloc(ACT * 4);
  }
  __bf16* qb    = (__bf16*)alloc(ACT * 2);
  __bf16* kb    = (__bf16*)alloc(ACT * 2);
  __bf16* vb    = (__bf16*)alloc(ACT * 2);
  __bf16* attnb = (__bf16*)alloc(ACT * 2);
  __bf16* o_b   = (__bf16*)alloc(ACT * 2);
  __bf16* h_b   = (__bf16*)alloc(ACT * 2);
  float*  o_f   = (float*)alloc(ACT * 4);
  float*  h_f   = (float*)alloc(ACT * 4);
  float*  proj  = (float*)alloc((size_t)M * 1024 * 4);
  (void)ws_size; (void)in_sizes; (void)n_in; (void)out_size;

  dim3 blk(256);
  // cast all weights to bf16 once per launch
  cast_f32_bf16<<<(QKVW + 255) / 256, blk, 0, stream>>>(qkv_w, qkvw_bf, (int)QKVW);
  cast_f32_bf16<<<(HWW + 255) / 256, blk, 0, stream>>>(hw_w, hww_bf, (int)HWW);

  for (int dir = 0; dir < 2; ++dir) {
    hipMemcpyAsync(x_f[dir], inputs, ACT * 4, hipMemcpyDeviceToDevice, stream);
    cast_f32_bf16<<<(ACT + 255) / 256, blk, 0, stream>>>(inputs, x_b[dir], (int)ACT);
  }

  auto gemm = [&](const __bf16* A, const __bf16* W, const float* bias,
                  float* Cf, __bf16* Cb, int N) {
    dim3 g(N / 64, M / 64);
    gemm_bf16_wmma<<<g, dim3(128), 0, stream>>>(A, W, bias, Cf, Cb, M, N, HID);
  };

  const int elemBlocks = (int)((ACT + 255) / 256);
  for (int l = 0; l < 2; ++l) {
    for (int dir = 0; dir < 2; ++dir) {
      const int ld = l * 2 + dir;
      const __bf16* Wq = qkvw_bf + (size_t)(ld * 4 + 0) * 512 * 512;
      const __bf16* Wk = qkvw_bf + (size_t)(ld * 4 + 1) * 512 * 512;
      const __bf16* Wv = qkvw_bf + (size_t)(ld * 4 + 2) * 512 * 512;
      const __bf16* Wo = qkvw_bf + (size_t)(ld * 4 + 3) * 512 * 512;
      const float* bq = qkv_b + (size_t)(ld * 4 + 0) * 512;
      const float* bk = qkv_b + (size_t)(ld * 4 + 1) * 512;
      const float* bv = qkv_b + (size_t)(ld * 4 + 2) * 512;
      const float* bo = qkv_b + (size_t)(ld * 4 + 3) * 512;
      const float* relp = rel + (size_t)ld * 8 * 34;

      hipMemcpyAsync(cachep[dir], x_f[dir], ACT * 4, hipMemcpyDeviceToDevice, stream);

      gemm(x_b[dir], Wq, bq, nullptr, qb, 512);
      gemm(x_b[dir], Wk, bk, nullptr, kb, 512);
      gemm(x_b[dir], Wv, bv, nullptr, vb, 512);

      // 2048 waves total: 64 q-tiles x 8 heads x 4 batches; 4 waves/block
      attn_banded<<<512, dim3(128), 0, stream>>>(qb, kb, vb, relp, attnb,
                                                 dir == 0 ? 1 : 0);

      gemm(attnb, Wo, bo, o_f, o_b, 512);

      // highway 0: proj = o @ W0^T + b0 ; gate with x = o_f -> h
      const __bf16* Wh0 = hww_bf + (size_t)(ld * 2 + 0) * 1024 * 512;
      const __bf16* Wh1 = hww_bf + (size_t)(ld * 2 + 1) * 1024 * 512;
      const float* bh0 = hw_b + (size_t)(ld * 2 + 0) * 1024;
      const float* bh1 = hw_b + (size_t)(ld * 2 + 1) * 1024;

      gemm(o_b, Wh0, bh0, proj, nullptr, 1024);
      highway_elem<<<elemBlocks, blk, 0, stream>>>(proj, o_f, h_f, h_b, (int)ACT);

      // highway 1: gate with x = h_f -> (o_f, o_b) reused as h2
      gemm(h_b, Wh1, bh1, proj, nullptr, 1024);
      highway_elem<<<elemBlocks, blk, 0, stream>>>(proj, h_f, o_f, o_b, (int)ACT);

      // residual (layers > 0) + update x + write output slice
      float* outbase = out + (size_t)l * 4096 * 1024 + (size_t)dir * 512;
      residual_out<<<elemBlocks, blk, 0, stream>>>(o_f, cachep[dir], l > 0 ? 1 : 0,
                                                   x_f[dir], x_b[dir], outbase,
                                                   (int)ACT);
    }
  }
}
